// SupplyChainGNN_88373247083004
// MI455X (gfx1250) — matrix operand order repaired and checked by reading, not compiled, not run on hardware
//
#include <hip/hip_runtime.h>

typedef __attribute__((ext_vector_type(16))) _Float16 v16h;
typedef __attribute__((ext_vector_type(4)))  _Float16 v4h;
typedef __attribute__((ext_vector_type(8)))  float    v8f;
typedef __attribute__((ext_vector_type(4)))  unsigned u32x4;
typedef __attribute__((ext_vector_type(4)))  int      i32x4;
typedef __attribute__((ext_vector_type(8)))  int      i32x8;

#define GNN_H 64
#define GNN_LAYERS 3

// ---------------------------------------------------------------- packed-A index
// Activation matrix is stored ONLY as f16 WMMA A-fragments:
//   hA[((mt*2 + ks)*32 + lane)*16 + e]
// where element e of lane (hi,m) holds A[mt*16+m][ks*32 + (e>>3)*16 + hi*8 + (e&7)].
// For channel block c0 (multiple of 4), the 4 channels map to 4 contiguous e.
__device__ inline size_t pack_idx(int n, int c0) {
  int mt = n >> 4, m = n & 15;
  int ks = c0 >> 5;
  int koff = c0 & 31;
  int group = koff >> 4;        // 0/1
  int r = koff & 15;            // 0,4,8,12
  int hi = r >> 3;              // 0/1
  int e0 = group * 8 + (r & 7); // first of 4 contiguous elements
  int lane = hi * 16 + m;
  return ((size_t)(mt * 2 + ks) * 32 + lane) * 16 + e0;
}

// ---------------------------------------------------------------- TDM: 1-D u16 tile -> LDS
__device__ inline void tdm_load_1d_u16(const void* gptr, void* lptr, int nelem) {
  unsigned long long ga = (unsigned long long)(uintptr_t)gptr;
  unsigned lds = (unsigned)(uintptr_t)lptr;   // low 32 bits of generic = LDS offset
  u32x4 g0 = { 1u,                                   // count=1, user descriptor
               lds,                                  // lds_addr
               (unsigned)ga,                         // global_addr[31:0]
               ((unsigned)(ga >> 32) & 0x01FFFFFFu) | (2u << 30) };  // addr[56:32] | type=2
  i32x8 g1 = { (int)(1u << 16),                              // data_size=1 (2B), mask=0
               (int)(((unsigned)nelem & 0xFFFFu) << 16),     // tensor_dim0[15:0]
               (int)((((unsigned)nelem >> 16) & 0xFFFFu) | (1u << 16)), // dim0 hi | tensor_dim1=1
               (int)(((unsigned)nelem & 0xFFFFu) << 16),     // tile_dim0 = nelem
               0,                                            // tile_dim1=0 (1-D), tile_dim2=0
               nelem,                                        // tensor_dim0_stride
               0, 0 };
  i32x4 gz = { 0, 0, 0, 0 };
#if defined(__clang_major__) && (__clang_major__ >= 23)
  i32x8 gz8 = { 0, 0, 0, 0, 0, 0, 0, 0 };
  __builtin_amdgcn_tensor_load_to_lds(g0, g1, gz, gz, gz8, 0);
#else
  __builtin_amdgcn_tensor_load_to_lds(g0, g1, gz, gz, 0);
#endif
}

// ---------------------------------------------------------------- degree / norm
__global__ void k_init_deg(float* __restrict__ deg, int n) {
  int i = blockIdx.x * blockDim.x + threadIdx.x;
  if (i < n) deg[i] = 1.0f;  // self loop
}

__global__ void k_degree(const int* __restrict__ col, float* __restrict__ deg, int E) {
  int e = blockIdx.x * blockDim.x + threadIdx.x;
  if (e < E) atomicAdd(&deg[col[e]], 1.0f);
}

__global__ void k_rsqrt_inplace(float* __restrict__ deg, int n) {
  int i = blockIdx.x * blockDim.x + threadIdx.x;
  if (i < n) deg[i] = rsqrtf(deg[i]);
}

// ---------------------------------------------------------------- encoder (writes packed f16 A)
__global__ void k_encoder(const float* __restrict__ x, const float* __restrict__ W,
                          const float* __restrict__ b, _Float16* __restrict__ hA, int N) {
  int tid = blockIdx.x * blockDim.x + threadIdx.x;  // N*16, 4 channels each
  if (tid >= N * 16) return;
  int n = tid >> 4;
  int c0 = (tid & 15) * 4;
  const float* xr = x + n * 5;
  float xv0 = xr[0], xv1 = xr[1], xv2 = xr[2], xv3 = xr[3], xv4 = xr[4];
  v4h o;
#pragma unroll
  for (int t = 0; t < 4; ++t) {
    int j = c0 + t;
    float acc = b[j];
    acc = fmaf(xv0, W[0 * GNN_H + j], acc);
    acc = fmaf(xv1, W[1 * GNN_H + j], acc);
    acc = fmaf(xv2, W[2 * GNN_H + j], acc);
    acc = fmaf(xv3, W[3 * GNN_H + j], acc);
    acc = fmaf(xv4, W[4 * GNN_H + j], acc);
    o[t] = (_Float16)fmaxf(acc, 0.0f);
  }
  *(v4h*)(hA + pack_idx(n, c0)) = o;
}

// ---------------------------------------------------------------- B fragment packing
// Bp[((nt*2+ks)*32 + lane)*16 + e] = (f16) W[(ks*32 + (e>>3)*16 + hi*8 + (e&7)) * nOut + nt*16 + m]
__global__ void k_pack_B(const float* __restrict__ W, _Float16* __restrict__ Bp,
                         int nOut, int nTiles) {
  int idx = blockIdx.x * blockDim.x + threadIdx.x;
  if (idx >= nTiles * 1024) return;
  int e    = idx & 15;
  int lane = (idx >> 4) & 31;
  int ks   = (idx >> 9) & 1;
  int nt   = idx >> 10;
  int m = lane & 15, hi = lane >> 4;
  int K = ks * 32 + (e >> 3) * 16 + hi * 8 + (e & 7);
  Bp[idx] = (_Float16)W[K * nOut + nt * 16 + m];
}

// ---------------------------------------------------------------- WMMA GEMM
// C[M x (NT*16)] = unpack(Ap) * unpack(Bp); M multiple of 16; NT compile-time.
// B staged through LDS by the Tensor Data Mover; each wave does 4 M-tiles.
template <int NT>
__global__ void k_wmma_gemm(const _Float16* __restrict__ Ap, const _Float16* __restrict__ Bp,
                            float* __restrict__ C, int mTiles) {
  __shared__ _Float16 ldsB[NT * 1024];  // NT n-tiles * 2 ks * 512 halves

  const int wave = threadIdx.x >> 5;
  const int lane = threadIdx.x & 31;

  if (wave == 0) {
    tdm_load_1d_u16(Bp, ldsB, NT * 1024);
    __builtin_amdgcn_s_wait_tensorcnt(0);
  }
  __syncthreads();

  // Work-unit id is identical across the wave: move it (and all derived
  // addressing) into SGPRs so tile guards become scalar branches.
  const int wu = __builtin_amdgcn_readfirstlane(
      blockIdx.x * (blockDim.x >> 5) + wave);
  const int mGroups = (mTiles + 3) >> 2;
  if (wu >= mGroups * NT) return;               // scalar branch
  const int nt = wu % NT;
  const int mtBase = (wu / NT) * 4;

  const v16h b0 = *(const v16h*)&ldsB[((nt * 2 + 0) * 32 + lane) * 16];
  const v16h b1 = *(const v16h*)&ldsB[((nt * 2 + 1) * 32 + lane) * 16];

  const int hi = lane >> 4, m = lane & 15;
  constexpr int LDC = NT * 16;

  const _Float16* Abase = Ap + (size_t)mtBase * 1024 + lane * 16;
  float* Cbase = C + (size_t)(mtBase * 16) * LDC + nt * 16 + m + (size_t)hi * 8 * LDC;

  if (mtBase + 4 <= mTiles) {
    // fast path: 4 full tiles, loads hoistable ahead of the WMMAs
#pragma unroll
    for (int i = 0; i < 4; ++i) {
      const v16h a0 = *(const v16h*)(Abase + (size_t)i * 1024);
      const v16h a1 = *(const v16h*)(Abase + (size_t)i * 1024 + 512);
      v8f c = {};
      c = __builtin_amdgcn_wmma_f32_16x16x32_f16(false, a0, false, b0, (short)0, c, false, false);
      c = __builtin_amdgcn_wmma_f32_16x16x32_f16(false, a1, false, b1, (short)0, c, false, false);
      float* Crow = Cbase + (size_t)i * 16 * LDC;
#pragma unroll
      for (int r = 0; r < 8; ++r) Crow[r * LDC] = c[r];
    }
  } else {
#pragma unroll
    for (int i = 0; i < 4; ++i) {
      if (mtBase + i < mTiles) {                // scalar branch
        const v16h a0 = *(const v16h*)(Abase + (size_t)i * 1024);
        const v16h a1 = *(const v16h*)(Abase + (size_t)i * 1024 + 512);
        v8f c = {};
        c = __builtin_amdgcn_wmma_f32_16x16x32_f16(false, a0, false, b0, (short)0, c, false, false);
        c = __builtin_amdgcn_wmma_f32_16x16x32_f16(false, a1, false, b1, (short)0, c, false, false);
        float* Crow = Cbase + (size_t)i * 16 * LDC;
#pragma unroll
        for (int r = 0; r < 8; ++r) Crow[r * LDC] = c[r];
      }
    }
  }
}

// ---------------------------------------------------------------- GCN aggregation
// agg[i,:] = hw[i,:] * dinv[i]^2  (self loop; fully initializes agg)
__global__ void k_agg_self(const float* __restrict__ hw, const float* __restrict__ dinv,
                           float* __restrict__ agg, int N) {
  int tid = blockIdx.x * blockDim.x + threadIdx.x;  // N*16 float4s
  if (tid >= N * 16) return;
  int n = tid >> 4;
  float s = dinv[n];
  s = s * s;
  float4 v = ((const float4*)hw)[tid];
  float4 o = { v.x * s, v.y * s, v.z * s, v.w * s };
  ((float4*)agg)[tid] = o;
}

// agg[col[e],:] += hw[row[e],:] * dinv[row]*dinv[col]; 4-deep grid-stride with prefetch
__global__ void k_scatter(const int* __restrict__ row, const int* __restrict__ col,
                          const float* __restrict__ dinv, const float* __restrict__ hw,
                          float* __restrict__ agg, int E, int T) {
  int tid = blockIdx.x * blockDim.x + threadIdx.x;
  if (tid >= T) return;
  const int total = E * 16;
  for (int j = tid; j < total; j += T) {
    int jn = j + T;
    if (jn < total) {                       // prefetch next gather row
      __builtin_prefetch(hw + (size_t)row[jn >> 4] * GNN_H, 0, 0);
    }
    int e = j >> 4, g = j & 15;
    int r = row[e];
    int cdst = col[e];
    float nrm = dinv[r] * dinv[cdst];
    float4 v = ((const float4*)(hw + (size_t)r * GNN_H))[g];
    float* dst = agg + (size_t)cdst * GNN_H + g * 4;
    atomicAdd(dst + 0, v.x * nrm);
    atomicAdd(dst + 1, v.y * nrm);
    atomicAdd(dst + 2, v.z * nrm);
    atomicAdd(dst + 3, v.w * nrm);
  }
}

// hA <- packed f16 relu(agg + b)
__global__ void k_bias_relu(const float* __restrict__ agg, const float* __restrict__ b,
                            _Float16* __restrict__ hA, int N) {
  int tid = blockIdx.x * blockDim.x + threadIdx.x;  // N*16
  if (tid >= N * 16) return;
  int n = tid >> 4;
  int c0 = (tid & 15) * 4;
  float4 v = ((const float4*)agg)[tid];
  float4 bb = *(const float4*)(b + c0);
  v4h o;
  o[0] = (_Float16)fmaxf(v.x + bb.x, 0.0f);
  o[1] = (_Float16)fmaxf(v.y + bb.y, 0.0f);
  o[2] = (_Float16)fmaxf(v.z + bb.z, 0.0f);
  o[3] = (_Float16)fmaxf(v.w + bb.w, 0.0f);
  *(v4h*)(hA + pack_idx(n, c0)) = o;
}

// ---------------------------------------------------------------- head tail
// out[n] = sum_j relu(hd[n,j] + b1[j]) * W2[j] + b2
__global__ void k_head_tail(const float* __restrict__ hd, const float* __restrict__ b1,
                            const float* __restrict__ W2, const float* __restrict__ b2,
                            float* __restrict__ out, int N) {
  int n = blockIdx.x * blockDim.x + threadIdx.x;
  if (n >= N) return;
  const float* r = hd + (size_t)n * 32;
  float acc = b2[0];
#pragma unroll
  for (int j = 0; j < 32; ++j) acc = fmaf(fmaxf(r[j] + b1[j], 0.0f), W2[j], acc);
  out[n] = acc;
}

// ---------------------------------------------------------------- launch
static inline int cdiv(int a, int b) { return (a + b - 1) / b; }

extern "C" void kernel_launch(void* const* d_in, const int* in_sizes, int n_in,
                              void* d_out, int out_size, void* d_ws, size_t ws_size,
                              hipStream_t stream) {
  const float* x      = (const float*)d_in[0];
  const int*   ei     = (const int*)d_in[1];
  const float* W_enc  = (const float*)d_in[2];
  const float* b_enc  = (const float*)d_in[3];
  const float* conv_W = (const float*)d_in[4];
  const float* conv_b = (const float*)d_in[5];
  const float* W_d1   = (const float*)d_in[6];
  const float* b_d1   = (const float*)d_in[7];
  const float* W_d2   = (const float*)d_in[8];
  const float* b_d2   = (const float*)d_in[9];
  const float* W_i1   = (const float*)d_in[10];
  const float* b_i1   = (const float*)d_in[11];
  const float* W_i2   = (const float*)d_in[12];
  const float* b_i2   = (const float*)d_in[13];

  const int N = in_sizes[0] / 5;
  const int E = in_sizes[1] / 2;
  const int* row = ei;
  const int* col = ei + E;
  const int mTiles = cdiv(N, 16);  // 3125 (N is 16-aligned)

  // workspace carve-out
  char* ws = (char*)d_ws;
  size_t off = 0;
  auto carve = [&](size_t bytes) -> void* {
    void* p = ws + off;
    off += (bytes + 255) & ~(size_t)255;
    return p;
  };
  float*    deg    = (float*)carve((size_t)N * 4);
  _Float16* hA     = (_Float16*)carve((size_t)mTiles * 1024 * 2);   // packed A fragments
  float*    hw     = (float*)carve((size_t)N * GNN_H * 4);
  float*    agg    = (float*)carve((size_t)N * GNN_H * 4);
  _Float16* BpConv = (_Float16*)carve((size_t)GNN_LAYERS * 4096 * 2);
  _Float16* BpD    = (_Float16*)carve((size_t)2048 * 2);
  _Float16* BpI    = (_Float16*)carve((size_t)2048 * 2);
  float*    hd     = hw;  // reuse: hw dead after last conv layer

  // pack weights into WMMA B-fragment order (every call; deterministic)
  for (int l = 0; l < GNN_LAYERS; ++l)
    k_pack_B<<<cdiv(4096, 256), 256, 0, stream>>>(conv_W + (size_t)l * GNN_H * GNN_H,
                                                  BpConv + (size_t)l * 4096, GNN_H, 4);
  k_pack_B<<<cdiv(2048, 256), 256, 0, stream>>>(W_d1, BpD, 32, 2);
  k_pack_B<<<cdiv(2048, 256), 256, 0, stream>>>(W_i1, BpI, 32, 2);

  // symmetric-norm degrees (with self loops)
  k_init_deg<<<cdiv(N, 256), 256, 0, stream>>>(deg, N);
  k_degree<<<cdiv(E, 256), 256, 0, stream>>>(col, deg, E);
  k_rsqrt_inplace<<<cdiv(N, 256), 256, 0, stream>>>(deg, N);

  // encoder -> packed A
  k_encoder<<<cdiv(N * 16, 256), 256, 0, stream>>>(x, W_enc, b_enc, hA, N);

  const int mGroups = cdiv(mTiles, 4);
  const int scatterT = cdiv(E * 16, 4);

  for (int l = 0; l < GNN_LAYERS; ++l) {
    k_wmma_gemm<4><<<cdiv(mGroups * 4, 8), 256, 0, stream>>>(
        hA, BpConv + (size_t)l * 4096, hw, mTiles);
    k_agg_self<<<cdiv(N * 16, 256), 256, 0, stream>>>(hw, deg, agg, N);
    k_scatter<<<cdiv(scatterT, 256), 256, 0, stream>>>(row, col, deg, hw, agg, E, scatterT);
    k_bias_relu<<<cdiv(N * 16, 256), 256, 0, stream>>>(agg, conv_b + (size_t)l * GNN_H, hA, N);
  }

  float* out = (float*)d_out;
  // demand head
  k_wmma_gemm<2><<<cdiv(mGroups * 2, 8), 256, 0, stream>>>(hA, BpD, hd, mTiles);
  k_head_tail<<<cdiv(N, 256), 256, 0, stream>>>(hd, b_d1, W_d2, b_d2, out, N);
  // inventory head
  k_wmma_gemm<2><<<cdiv(mGroups * 2, 8), 256, 0, stream>>>(hA, BpI, hd, mTiles);
  k_head_tail<<<cdiv(N, 256), 256, 0, stream>>>(hd, b_i1, W_i2, b_i2, out + N, N);
}
